// LFQ_VAE_25409026523970
// MI455X (gfx1250) — matrix operand order repaired and checked by reading, not compiled, not run on hardware
//
#include <hip/hip_runtime.h>

typedef __attribute__((ext_vector_type(16))) _Float16 v16h;
typedef __attribute__((ext_vector_type(8)))  float    v8f;

#define WAVES 8
#define MT    128          // tokens per workgroup (8 waves x 16 rows)
#define NTOK  (8*4096)
#define FDIM  256
#define H1    64
#define HD    128
#define LDIM  64
#define NC    1024

// ---- d_ws byte layout (W2..Wo contiguous so they stage to LDS in one copy) ----
#define OFF_W1   0u          // 64x256  f16 -> 32768   (streamed from global)
#define OFF_W2   32768u      // 128x64  f16 -> 16384   (LDS)
#define OFF_WL   49152u      // 64x128  f16 -> 16384   (LDS)
#define OFF_CB   65536u      // 1024x64 f16 -> 131072  (LDS)
#define OFF_WD1  196608u     // 64x64   f16 -> 8192    (LDS)
#define OFF_WD2  204800u     // 128x64  f16 -> 16384   (LDS)
#define OFF_WO   221184u     // 256x128 f16 -> 65536   (LDS)
#define OFF_CSQ  286720u     // 1024 f32    -> 4096    (LDS)
#define OFF_ACC  290816u     // 2 f32 accumulators

#define LDSW_BYTES 253952u   // W2..Wo staged block (15,872 x 16B = 62 chunks/thread)
#define LDS_TOTAL  258048u   // + csq (256 x 16B = 1 chunk/thread)

__device__ __forceinline__ v8f wmma16(v16h a, v16h b, v8f c) {
  // D = A(16x32,f16) x B(32x16,f16) + C(16x16,f32)
  return __builtin_amdgcn_wmma_f32_16x16x32_f16(false, a, false, b, (short)0, c,
                                                false, false);
}
__device__ __forceinline__ float gelu_f(float v) {
  return 0.5f * v * (1.0f + erff(v * 0.70710678118654752440f));
}
// A-fragment gather from f16 row-major storage; p = row_base + kt*32 + half*16
__device__ __forceinline__ v16h afrag(const _Float16* p) {
  v16h a;
#pragma unroll
  for (int e = 0; e < 8; ++e) { a[e] = p[e]; a[e + 8] = p[e + 8]; }
  return a;
}
// async DMA of one 16B chunk: global -> LDS, tracked by ASYNCcnt (no VGPR round-trip)
__device__ __forceinline__ void async_g2l_b128(uint32_t lds_off, const void* gptr) {
  uint64_t ga = (uint64_t)(uintptr_t)gptr;
  asm volatile("global_load_async_to_lds_b128 %0, %1, off"
               :: "v"(lds_off), "v"(ga) : "memory");
}

// ---------------- prep kernels ----------------
__global__ void k_cvt(const float* __restrict__ s, _Float16* __restrict__ d, int n) {
  int i = blockIdx.x * blockDim.x + threadIdx.x;
  if (i < n) d[i] = (_Float16)s[i];
}
__global__ void k_csq(const float* __restrict__ cb, float* __restrict__ csq) {
  int c = blockIdx.x * blockDim.x + threadIdx.x;
  if (c < NC) {
    float s = 0.f;
    for (int k = 0; k < LDIM; ++k) { float v = cb[c * LDIM + k]; s += v * v; }
    csq[c] = s;
  }
}
__global__ void k_zero(float* a) { if (threadIdx.x < 2) a[threadIdx.x] = 0.f; }
__global__ void k_final(const float* __restrict__ acc, float* __restrict__ loss) {
  if (threadIdx.x == 0)
    loss[0] = acc[0] / (float)((size_t)NTOK * FDIM) +
              0.5f * (acc[1] / (float)((size_t)NTOK * LDIM));
}

// ---------------- fused VQ-VAE forward ----------------
__global__ __launch_bounds__(256) void k_vqvae(
    const float* __restrict__ x,
    const _Float16* __restrict__ W1h, const float* __restrict__ b1,
    const _Float16* __restrict__ W2h,   // base of contiguous W2..Wo staged block
    const float* __restrict__ b2,
    const float* __restrict__ bl,
    const float* __restrict__ cb32,
    const float* __restrict__ csq,
    const float* __restrict__ bd1, const float* __restrict__ bd2,
    const float* __restrict__ bo,
    float* __restrict__ zq_out, float* __restrict__ accum)
{
  // 258,048B staged weights + 49,152B activation buffers + 1,536B misc = 308,736B
  __shared__ __align__(64) char    s_w[LDS_TOTAL];
  __shared__ __align__(32) _Float16 s_big[WAVES][16][HD];     // h2 / r2
  __shared__ __align__(32) _Float16 s_small[WAVES][16][LDIM]; // h1 / ze16 / r1
  __shared__ int   s_idx[WAVES][16];
  __shared__ float s_red[256];

  const int tid  = threadIdx.x;
  const int wave = tid >> 5;
  const int lane = tid & 31;
  const int nl   = lane & 15;   // lane's M (A) / N (B,C) coordinate
  const int half = lane >> 4;   // K-half selector
  const int rb   = blockIdx.x * MT + wave * 16;   // first token row of this wave

  // ===== kick off async staging: W2,Wl,CB,Wd1,Wd2,Wo + csq -> LDS (ASYNCcnt) =====
  {
    const uint32_t lds_base = (uint32_t)(uintptr_t)(void*)s_w;  // LDS byte offset
#pragma unroll 1
    for (int i = tid; i < (int)(LDSW_BYTES / 16); i += 256)     // 62 iters/thread
      async_g2l_b128(lds_base + (uint32_t)i * 16u, (const char*)W2h + (size_t)i * 16u);
    async_g2l_b128(lds_base + LDSW_BYTES + (uint32_t)tid * 16u, // csq: 1 chunk/thread
                   (const char*)csq + (size_t)tid * 16u);
  }

  _Float16* bufS = &s_small[wave][0][0];  // stride LDIM
  _Float16* bufB = &s_big[wave][0][0];    // stride HD

  // ===== stage 1 (overlaps staging DMA): h1 = gelu(x @ W1^T + b1)  K=256 -> N=64
  {
    v8f acc[4] = {};
    const float* xrow = x + (size_t)(rb + nl) * FDIM;
#pragma unroll
    for (int kt = 0; kt < 8; ++kt) {
      const float* p = xrow + kt * 32 + half * 16;
      v16h a;
#pragma unroll
      for (int e = 0; e < 8; ++e) { a[e] = (_Float16)p[e]; a[e + 8] = (_Float16)p[e + 8]; }
#pragma unroll
      for (int nt = 0; nt < 4; ++nt) {
        v16h b = *(const v16h*)(W1h + (nt * 16 + nl) * FDIM + kt * 32 + half * 16);
        acc[nt] = wmma16(a, b, acc[nt]);
      }
    }
#pragma unroll
    for (int nt = 0; nt < 4; ++nt) {
      float bias = b1[nt * 16 + nl];
#pragma unroll
      for (int r = 0; r < 8; ++r) {
        float v = gelu_f(acc[nt][r] + bias);
        bufS[(r + 8 * half) * LDIM + nt * 16 + nl] = (_Float16)v;
      }
    }
  }
  // ===== drain DMA + block barrier before anyone reads staged LDS =====
  asm volatile("s_wait_asynccnt 0x0" ::: "memory");
  __syncthreads();
  const _Float16* lW2  = (const _Float16*)(s_w + (OFF_W2  - OFF_W2));
  const _Float16* lWl  = (const _Float16*)(s_w + (OFF_WL  - OFF_W2));
  const _Float16* lCB  = (const _Float16*)(s_w + (OFF_CB  - OFF_W2));
  const _Float16* lWd1 = (const _Float16*)(s_w + (OFF_WD1 - OFF_W2));
  const _Float16* lWd2 = (const _Float16*)(s_w + (OFF_WD2 - OFF_W2));
  const _Float16* lWo  = (const _Float16*)(s_w + (OFF_WO  - OFF_W2));
  const float*    lcsq = (const float*)(s_w + LDSW_BYTES);

  // ===== stage 2: h2 = gelu(h1 @ W2^T + b2)   K=64 -> N=128  (W2 from LDS)
  {
    v8f acc[8] = {};
#pragma unroll
    for (int kt = 0; kt < 2; ++kt) {
      v16h a = afrag(bufS + nl * LDIM + kt * 32 + half * 16);
#pragma unroll
      for (int nt = 0; nt < 8; ++nt) {
        v16h b = *(const v16h*)(lW2 + (nt * 16 + nl) * H1 + kt * 32 + half * 16);
        acc[nt] = wmma16(a, b, acc[nt]);
      }
    }
#pragma unroll
    for (int nt = 0; nt < 8; ++nt) {
      float bias = b2[nt * 16 + nl];
#pragma unroll
      for (int r = 0; r < 8; ++r) {
        float v = gelu_f(acc[nt][r] + bias);
        bufB[(r + 8 * half) * HD + nt * 16 + nl] = (_Float16)v;
      }
    }
  }
  // ===== stage 3: z_e = h2 @ Wl^T + bl   K=128 -> N=64  (Wl from LDS)
  {
    v8f acc[4] = {};
#pragma unroll
    for (int kt = 0; kt < 4; ++kt) {
      v16h a = afrag(bufB + nl * HD + kt * 32 + half * 16);
#pragma unroll
      for (int nt = 0; nt < 4; ++nt) {
        v16h b = *(const v16h*)(lWl + (nt * 16 + nl) * HD + kt * 32 + half * 16);
        acc[nt] = wmma16(a, b, acc[nt]);
      }
    }
#pragma unroll
    for (int nt = 0; nt < 4; ++nt) {
      float bias = bl[nt * 16 + nl];
#pragma unroll
      for (int r = 0; r < 8; ++r) {
        int m = r + 8 * half;
        float v = acc[nt][r] + bias;
        bufS[m * LDIM + nt * 16 + nl] = (_Float16)v;            // ze16 (A for distances)
        zq_out[(size_t)(rb + m) * LDIM + nt * 16 + nl] = v;     // z_e f32, staged in d_out
      }
    }
  }
  // ===== nearest codebook: argmin_c (||C||^2 - 2 z.C), codebook from LDS =====
  {
    v16h a0 = afrag(bufS + nl * LDIM + half * 16);
    v16h a1 = afrag(bufS + nl * LDIM + 32 + half * 16);
    float bs[8]; int bi[8];
#pragma unroll
    for (int r = 0; r < 8; ++r) { bs[r] = 3.4e38f; bi[r] = 0; }
    for (int nt = 0; nt < NC / 16; ++nt) {
      v8f acc = {};
      v16h c0 = *(const v16h*)(lCB + (nt * 16 + nl) * LDIM + half * 16);
      v16h c1 = *(const v16h*)(lCB + (nt * 16 + nl) * LDIM + 32 + half * 16);
      acc = wmma16(a0, c0, acc);
      acc = wmma16(a1, c1, acc);
      float cs = lcsq[nt * 16 + nl];
      int   ci = nt * 16 + nl;
#pragma unroll
      for (int r = 0; r < 8; ++r) {
        float s = cs - 2.0f * acc[r];
        if (s < bs[r]) { bs[r] = s; bi[r] = ci; }
      }
    }
    // cross-lane argmin within each 16-lane half (rows r+8*half), lowest index on tie
#pragma unroll
    for (int m = 1; m < 16; m <<= 1) {
#pragma unroll
      for (int r = 0; r < 8; ++r) {
        float os = __shfl_xor(bs[r], m, 32);
        int   oi = __shfl_xor(bi[r], m, 32);
        if (os < bs[r] || (os == bs[r] && oi < bi[r])) { bs[r] = os; bi[r] = oi; }
      }
    }
    if (nl == 0) {
#pragma unroll
      for (int r = 0; r < 8; ++r) s_idx[wave][r + 8 * half] = bi[r];
    }
  }
  // ===== z_q scatter (overwrite z_e in d_out) + commitment partial =====
  float commit = 0.f;
#pragma unroll 1
  for (int i = 0; i < 16; ++i) {
    int c = s_idx[wave][i];
    size_t ro = (size_t)(rb + i) * LDIM;
    float q0 = cb32[c * LDIM + lane];
    float q1 = cb32[c * LDIM + lane + 32];
    float e0 = zq_out[ro + lane];
    float e1 = zq_out[ro + lane + 32];
    zq_out[ro + lane]      = q0;
    zq_out[ro + lane + 32] = q1;
    float d0 = q0 - e0, d1 = q1 - e1;
    commit += d0 * d0 + d1 * d1;
  }
  // ===== decoder 1: r1 = gelu(z_q @ Wd1^T + bd1)   K=64 -> N=64  (LDS) =====
  {
    v8f acc[4] = {};
    const _Float16* zrow = lCB + s_idx[wave][nl] * LDIM;  // lane's z_q row (f16, LDS)
#pragma unroll
    for (int kt = 0; kt < 2; ++kt) {
      v16h a = afrag(zrow + kt * 32 + half * 16);
#pragma unroll
      for (int nt = 0; nt < 4; ++nt) {
        v16h b = *(const v16h*)(lWd1 + (nt * 16 + nl) * LDIM + kt * 32 + half * 16);
        acc[nt] = wmma16(a, b, acc[nt]);
      }
    }
#pragma unroll
    for (int nt = 0; nt < 4; ++nt) {
      float bias = bd1[nt * 16 + nl];
#pragma unroll
      for (int r = 0; r < 8; ++r) {
        float v = gelu_f(acc[nt][r] + bias);
        bufS[(r + 8 * half) * LDIM + nt * 16 + nl] = (_Float16)v;   // r1 (reuses ze16)
      }
    }
  }
  // ===== decoder 2: r2 = gelu(r1 @ Wd2^T + bd2)   K=64 -> N=128  (LDS) =====
  {
    v8f acc[8] = {};
#pragma unroll
    for (int kt = 0; kt < 2; ++kt) {
      v16h a = afrag(bufS + nl * LDIM + kt * 32 + half * 16);
#pragma unroll
      for (int nt = 0; nt < 8; ++nt) {
        v16h b = *(const v16h*)(lWd2 + (nt * 16 + nl) * LDIM + kt * 32 + half * 16);
        acc[nt] = wmma16(a, b, acc[nt]);
      }
    }
#pragma unroll
    for (int nt = 0; nt < 8; ++nt) {
      float bias = bd2[nt * 16 + nl];
#pragma unroll
      for (int r = 0; r < 8; ++r) {
        float v = gelu_f(acc[nt][r] + bias);
        bufB[(r + 8 * half) * HD + nt * 16 + nl] = (_Float16)v;     // r2 (reuses h2)
      }
    }
  }
  // ===== output: x_recon = r2 @ Wo^T + bo  K=128 -> N=256 (Wo from LDS) =====
  float recon = 0.f;
  {
    v16h af4[4];
#pragma unroll
    for (int kt = 0; kt < 4; ++kt) af4[kt] = afrag(bufB + nl * HD + kt * 32 + half * 16);
#pragma unroll 2
    for (int nt = 0; nt < 16; ++nt) {
      v8f acc = {};
#pragma unroll
      for (int kt = 0; kt < 4; ++kt) {
        v16h b = *(const v16h*)(lWo + (nt * 16 + nl) * HD + kt * 32 + half * 16);
        acc = wmma16(af4[kt], b, acc);
      }
      float bias = bo[nt * 16 + nl];
#pragma unroll
      for (int r = 0; r < 8; ++r) {
        int m = r + 8 * half;
        float v  = acc[r] + bias;
        float xv = x[(size_t)(rb + m) * FDIM + nt * 16 + nl];
        float d  = v - xv;
        recon += d * d;
      }
    }
  }
  // ===== block reduction of loss partials -> global accumulators =====
  s_red[tid] = recon;
  __syncthreads();
  for (int s = 128; s > 0; s >>= 1) { if (tid < s) s_red[tid] += s_red[tid + s]; __syncthreads(); }
  if (tid == 0) atomicAdd(&accum[0], s_red[0]);
  __syncthreads();
  s_red[tid] = commit;
  __syncthreads();
  for (int s = 128; s > 0; s >>= 1) { if (tid < s) s_red[tid] += s_red[tid + s]; __syncthreads(); }
  if (tid == 0) atomicAdd(&accum[1], s_red[0]);
}

extern "C" void kernel_launch(void* const* d_in, const int* in_sizes, int n_in,
                              void* d_out, int out_size, void* d_ws, size_t ws_size,
                              hipStream_t stream) {
  (void)in_sizes; (void)n_in; (void)out_size; (void)ws_size;
  const float* x    = (const float*)d_in[0];
  const float* W1   = (const float*)d_in[1];
  const float* b1   = (const float*)d_in[2];
  const float* W2   = (const float*)d_in[3];
  const float* b2   = (const float*)d_in[4];
  const float* Wl   = (const float*)d_in[5];
  const float* bl   = (const float*)d_in[6];
  const float* cb32 = (const float*)d_in[7];
  const float* Wd1  = (const float*)d_in[8];
  const float* bd1  = (const float*)d_in[9];
  const float* Wd2  = (const float*)d_in[10];
  const float* bd2  = (const float*)d_in[11];
  const float* Wo   = (const float*)d_in[12];
  const float* bo   = (const float*)d_in[13];

  char* ws = (char*)d_ws;
  _Float16* W1h  = (_Float16*)(ws + OFF_W1);
  _Float16* W2h  = (_Float16*)(ws + OFF_W2);
  _Float16* Wlh  = (_Float16*)(ws + OFF_WL);
  _Float16* CBh  = (_Float16*)(ws + OFF_CB);
  _Float16* Wd1h = (_Float16*)(ws + OFF_WD1);
  _Float16* Wd2h = (_Float16*)(ws + OFF_WD2);
  _Float16* Woh  = (_Float16*)(ws + OFF_WO);
  float*    csq  = (float*)(ws + OFF_CSQ);
  float*    acc  = (float*)(ws + OFF_ACC);

  auto cvt = [&](const float* s, _Float16* d, int n) {
    k_cvt<<<(n + 255) / 256, 256, 0, stream>>>(s, d, n);
  };
  cvt(W1,   W1h,  64 * 256);
  cvt(W2,   W2h,  128 * 64);
  cvt(Wl,   Wlh,  64 * 128);
  cvt(cb32, CBh,  NC * LDIM);
  cvt(Wd1,  Wd1h, 64 * 64);
  cvt(Wd2,  Wd2h, 128 * 64);
  cvt(Wo,   Woh,  256 * 128);
  k_csq<<<(NC + 255) / 256, 256, 0, stream>>>(cb32, csq);
  k_zero<<<1, 64, 0, stream>>>(acc);

  k_vqvae<<<NTOK / MT, 256, 0, stream>>>(x, W1h, b1, W2h, b2, bl, cb32, csq,
                                         bd1, bd2, bo, (float*)d_out, acc);

  k_final<<<1, 32, 0, stream>>>(acc, (float*)d_out + (size_t)NTOK * LDIM);
}
